// SimpleTOFPredictor_36515811951008
// MI455X (gfx1250) — matrix operand order repaired and checked by reading, not compiled, not run on hardware
//
#include <hip/hip_runtime.h>

typedef __attribute__((ext_vector_type(16))) _Float16 v16h;
typedef __attribute__((ext_vector_type(8)))  float    v8f;
typedef __attribute__((ext_vector_type(4)))  int      v4i;

#define NB1    4
#define NB2    1024
#define NRCV   64
#define TLEN   16384
#define CROP   256
#define NCHUNK 9          // ceil((CROP + 16 lags) / 32) K-chunks
#define EPAD   16         // zeros before echo (shifts reach index -15)
#define ELEN   320        // 16 pad + 256 echo + 48 pad (shifts reach 303)

#define SMEM_STG   (TLEN * 4)   // f32 async staging buffer
#define SMEM_REC   (TLEN * 2)   // f16 recording row
#define SMEM_ECHO  (ELEN * 2)   // padded f16 echo
#define SMEM_TOTAL (SMEM_STG + SMEM_REC + SMEM_ECHO)

#define AS1 __attribute__((address_space(1)))
#define AS3 __attribute__((address_space(3)))

__constant__ float kScale = 96000.0f / 343.0f;

__global__ __launch_bounds__(256) void tof_corr_wmma(
    const float* __restrict__ recordings,   // (4, 64, 16384)
    const float* __restrict__ samples,      // (4, 1024, 3)
    const float* __restrict__ emitter,      // (3,)
    const float* __restrict__ receivers,    // (64, 3)
    const float* __restrict__ echo,         // (256,)
    float* __restrict__ out)                // (4, 1024), pre-zeroed
{
    extern __shared__ char smem[];
    float*    stg     = (float*)smem;
    _Float16* sh_rec  = (_Float16*)(smem + SMEM_STG);
    _Float16* sh_echo = (_Float16*)(smem + SMEM_STG + SMEM_REC);

    const int b1  = blockIdx.x / NRCV;
    const int r   = blockIdx.x % NRCV;
    const int tid = threadIdx.x;

    const float* row = recordings + ((size_t)b1 * NRCV + r) * TLEN;

    // ---- zero echo pads + load echo (disjoint index ranges, no race) ----
    if (tid < EPAD)               sh_echo[tid] = (_Float16)0.0f;
    if (tid < ELEN - EPAD - CROP) sh_echo[EPAD + CROP + tid] = (_Float16)0.0f;
    sh_echo[EPAD + tid] = (_Float16)echo[tid];   // tid < 256 == CROP

    // ---- stage recording row: async DMA f32 -> LDS, then convert to f16 ----
#if __has_builtin(__builtin_amdgcn_global_load_async_to_lds_b128)
    #pragma unroll
    for (int i = 0; i < TLEN / (256 * 4); ++i) {        // 16 x b128 per thread
        const int e = (i * 256 + tid) * 4;
        __builtin_amdgcn_global_load_async_to_lds_b128(
            (AS1 v4i*)(row + e), (AS3 v4i*)(stg + e), 0, 0);
    }
#if __has_builtin(__builtin_amdgcn_s_wait_asynccnt)
    __builtin_amdgcn_s_wait_asynccnt(0);
#else
    asm volatile("s_wait_asynccnt 0x0" ::: "memory");
#endif
    __syncthreads();
    for (int i = tid * 4; i < TLEN; i += 256 * 4) {
        float4 v = *(const float4*)(stg + i);
        sh_rec[i + 0] = (_Float16)v.x;
        sh_rec[i + 1] = (_Float16)v.y;
        sh_rec[i + 2] = (_Float16)v.z;
        sh_rec[i + 3] = (_Float16)v.w;
    }
#else
    for (int i = tid * 4; i < TLEN; i += 256 * 4) {
        float4 v = *(const float4*)(row + i);
        sh_rec[i + 0] = (_Float16)v.x;
        sh_rec[i + 1] = (_Float16)v.y;
        sh_rec[i + 2] = (_Float16)v.z;
        sh_rec[i + 3] = (_Float16)v.w;
    }
#endif
    __syncthreads();

    const float ex = emitter[0], ey = emitter[1], ez = emitter[2];
    const float rx = receivers[r * 3 + 0];
    const float ry = receivers[r * 3 + 1];
    const float rz = receivers[r * 3 + 2];

    const int lane = tid & 31;
    const int n    = lane & 15;               // A row / B column / q-slot
    const int koff = (lane & 16) ? 16 : 0;    // A layout K offset
    const int kb   = (lane & 16) ? 16 : 0;    // B layout K base

    // ---- A = shifted echo, constant across all tiles: built once ----
    // A[m,k] = echo_pad[c*32 + k - m]  =>  D[m,n] = C[i0_n + m]
    v16h Aecho[NCHUNK];
    #pragma unroll
    for (int c = 0; c < NCHUNK; ++c) {
        const int base = EPAD + c * 32 + koff - n;
        #pragma unroll
        for (int j = 0; j < 8; ++j) {
            Aecho[c][j]     = sh_echo[base + j];
            Aecho[c][j + 8] = sh_echo[base + 8 + j];
        }
    }

    const int wave = tid >> 5;
    for (int tile = wave; tile < NB2 / 16; tile += 8) {
        const int q = tile * 16 + n;

        // geometry -> window base + constant lerp weights (one per q)
        const float* s = samples + ((size_t)b1 * NB2 + q) * 3;
        float sx = s[0], sy = s[1], sz = s[2];
        float dx = sx - ex, dy = sy - ey, dz = sz - ez;
        float d_es = sqrtf(dx * dx + dy * dy + dz * dz);
        dx = sx - rx; dy = sy - ry; dz = sz - rz;
        float d_sr = sqrtf(dx * dx + dy * dy + dz * dz);

        float start = (d_es + d_sr) * kScale;
        float fl    = floorf(start);
        float f     = start - fl;
        float w0    = 1.0f - f;
        int   ib    = (int)fl;
        ib = ib < 0 ? 0 : (ib > (TLEN - 32 * NCHUNK) ? (TLEN - 32 * NCHUNK) : ib);

        const _Float16* wbase = sh_rec + ib + kb;
        auto loadB = [&](v16h& b, int c) {
            const _Float16* wp = wbase + c * 32;
            #pragma unroll
            for (int j = 0; j < 16; ++j) b[j] = wp[j];
        };

        v8f acc;
        #pragma unroll
        for (int v = 0; v < 8; ++v) acc[v] = 0.0f;

        // ---- 9 chunks, one-deep software pipeline:
        //      prefetch B(c+1) before WMMA(c) so LDS latency overlaps math ----
        v16h bcur, bnxt;
        loadB(bcur, 0);
        #pragma unroll
        for (int c = 0; c < NCHUNK; ++c) {
            if (c + 1 < NCHUNK) loadB(bnxt, c + 1);
            acc = __builtin_amdgcn_wmma_f32_16x16x32_f16(
                false, Aecho[c], false, bcur, (short)0, acc, false, false);
            bcur = bnxt;
        }

        // lanes 0-15: acc[0]=C[i0], acc[1]=C[i0+1] for this lane's own q
        if (lane < 16) {
            float val = w0 * acc[0] + f * acc[1];
            val = val > 0.0f ? val : 0.0f;           // ReLU before r-sum
            unsafeAtomicAdd(out + (size_t)b1 * NB2 + q, val);
        }
    }
}

extern "C" void kernel_launch(void* const* d_in, const int* in_sizes, int n_in,
                              void* d_out, int out_size, void* d_ws, size_t ws_size,
                              hipStream_t stream) {
    const float* recordings = (const float*)d_in[0];
    const float* samples    = (const float*)d_in[1];
    const float* emitter    = (const float*)d_in[2];
    const float* receivers  = (const float*)d_in[3];
    const float* echo       = (const float*)d_in[4];
    float*       out        = (float*)d_out;

    // atomics accumulate over receivers -> zero the output every call
    (void)hipMemsetAsync(out, 0, (size_t)out_size * sizeof(float), stream);

    dim3 grid(NB1 * NRCV);   // one workgroup per (b1, receiver)
    dim3 block(256);         // 8 waves of 32
    tof_corr_wmma<<<grid, block, SMEM_TOTAL, stream>>>(
        recordings, samples, emitter, receivers, echo, out);
}
